// RoPEAttention_71442486001863
// MI455X (gfx1250) — matrix operand order repaired and checked by reading, not compiled, not run on hardware
//
#include <hip/hip_runtime.h>
#include <hip/hip_bf16.h>
#include <math.h>

// ---------------------------------------------------------------------------
// Problem constants (match reference)
// ---------------------------------------------------------------------------
#define DIMX   1024
#define HEADS  16
#define DH     64
#define INNER  1024          // HEADS*DH
#define QKV3   3072          // 3*INNER
#define NSEQ   2048
#define NBATCH 2
#define NROWS  4096          // NBATCH*NSEQ
#define SCALE  0.125f        // DH^-0.5

typedef __attribute__((ext_vector_type(16))) __bf16    v16bf;
typedef __attribute__((ext_vector_type(8)))  float     v8f;
typedef __attribute__((ext_vector_type(4)))  unsigned  u32x4;  // trivial uint4
typedef __attribute__((ext_vector_type(2)))  float     f32x2;  // trivial float2

union FragV { u32x4 q[2]; unsigned u[8]; v16bf v; };

__device__ __forceinline__ unsigned short f2bf(float f) {
  unsigned u = __builtin_bit_cast(unsigned, f);
  u += 0x7FFFu + ((u >> 16) & 1u);          // round-to-nearest-even
  return (unsigned short)(u >> 16);
}
__device__ __forceinline__ unsigned pack2(unsigned short lo, unsigned short hi) {
  return (unsigned)lo | ((unsigned)hi << 16);
}
__device__ __forceinline__ v8f wmma_bf16(v16bf a, v16bf b, v8f c) {
  return __builtin_amdgcn_wmma_f32_16x16x32_bf16(false, a, false, b, (short)0, c,
                                                 false, false);
}

// CDNA5 async global->LDS copy (tracked by ASYNCcnt). lds_off = raw LDS byte
// offset (= low 32 bits of the flat shared address, per ISA aperture rules).
__device__ __forceinline__ void async_b128(unsigned lds_off, const void* gaddr) {
  asm volatile("global_load_async_to_lds_b128 %0, %1, off"
               :: "v"(lds_off), "v"(gaddr) : "memory");
}
__device__ __forceinline__ void wait_async0() {
  asm volatile("s_wait_asynccnt 0x0" ::: "memory");
}
__device__ __forceinline__ void wait_async8() {
  asm volatile("s_wait_asynccnt 0x8" ::: "memory");
}

// ---------------------------------------------------------------------------
// Kernel A: fp32 -> bf16 weight conversion with fragment-native swizzle.
// Tile = 32(K) x 64(N). Within a tile, dword index = (nc*32 + lane)*8 + j,
// holding elements B[k][n] with n = nc*16 + lane%16, k = (lane/16)*16 + 2j(+1).
// Tiles ordered (ct*32 + kt). Each lane's fragment = 32 contiguous bytes.
// ---------------------------------------------------------------------------
__global__ void swizzle_w_kernel(const float* __restrict__ src,
                                 unsigned short* __restrict__ dst, int ncols) {
  const int k = blockIdx.y;                       // 0..1023
  const int c = blockIdx.x * 256 + threadIdx.x;   // 0..ncols-1
  const float v = src[(size_t)k * ncols + c];
  const int kt = k >> 5, kl = k & 31;
  const int ct = c >> 6, cl = c & 63;
  const int nc = cl >> 4, nl = cl & 15;
  const int hs = kl >> 4, j = (kl & 15) >> 1, h = kl & 1;
  const int lane = hs * 16 + nl;
  const size_t di = (((size_t)(ct * 32 + kt)) * 1024 +
                     (size_t)(nc * 32 + lane) * 8 + j) * 2 + h;
  dst[di] = f2bf(v);
}

// ---------------------------------------------------------------------------
// Kernel B: QKV GEMM (X[4096x1024] @ Wqkv[1024x3072]) + fused RoPE epilogue.
// 128 threads (4 waves); block tile 64(M) x 64(N); wave = 16 rows x 64 cols.
// A staged in LDS (packed bf16 dwords), fragments read as 2x b128.
// B read straight from pre-swizzled global weights as 2x b128 (L2-resident).
// Writes Q,K head-major row-major; V transposed (bh, d, n) for attention.
// ---------------------------------------------------------------------------
__global__ __launch_bounds__(128) void qkv_rope_kernel(
    const float* __restrict__ X, const unsigned short* __restrict__ Wsw,
    unsigned short* __restrict__ Qb, unsigned short* __restrict__ Kb,
    unsigned short* __restrict__ Vt) {
  __shared__ unsigned short Xs[64][40];   // 64 x 32, 80B rows (16B aligned)

  const int tid = threadIdx.x;
  const int lane = tid & 31, w = tid >> 5;
  const int nl = lane & 15, hs = lane >> 4;
  const int ct = blockIdx.x;              // 64-col tile
  const int m0 = blockIdx.y * 64;
  const int mrow = w * 16 + nl;
  const u32x4* __restrict__ W4 = (const u32x4*)Wsw;

  v8f acc[4] = {};

  for (int kt = 0; kt < DIMX / 32; ++kt) {
    // stage X tile as packed bf16 dwords (8 per thread)
    #pragma unroll
    for (int i = 0; i < 8; ++i) {
      int idx = tid + i * 128;
      int r = idx >> 4, cd = idx & 15;
      const f32x2 f2 = *(const f32x2*)&X[(size_t)(m0 + r) * DIMX + kt * 32 + cd * 2];
      *(unsigned*)&Xs[r][cd * 2] = pack2(f2bf(f2.x), f2bf(f2.y));
    }
    __syncthreads();

    FragV a;
    a.q[0] = *(const u32x4*)&Xs[mrow][hs * 16];
    a.q[1] = *(const u32x4*)&Xs[mrow][hs * 16 + 8];

    const size_t tb = ((size_t)(ct * 32 + kt)) * 256;   // u32x4 units per tile
    #pragma unroll
    for (int nc = 0; nc < 4; ++nc) {
      FragV b;
      const u32x4* wp = W4 + tb + (size_t)(nc * 32 + lane) * 2;
      b.q[0] = wp[0];
      b.q[1] = wp[1];
      acc[nc] = wmma_bf16(a.v, b.v, acc[nc]);
    }
    __syncthreads();
  }

  // epilogue: RoPE for q/k, scatter to attention buffers
  const int c0 = ct * 64;
  const int qkv = c0 >> 10;               // 0=q 1=k 2=v
  const int head = (c0 & 1023) >> 6;

  if (qkv < 2) {
    #pragma unroll
    for (int r = 0; r < 8; ++r) {
      const int grow = m0 + w * 16 + r + hs * 8;
      const float npos = (float)(grow & (NSEQ - 1));
      #pragma unroll
      for (int nc = 0; nc < 2; ++nc) {
        const int d = nc * 16 + nl;                  // 0..31
        const float theta = __expf((float)d * -0.28782313662425572f);
        const float ang = npos * theta;
        const float cs = __cosf(ang), sn = __sinf(ang);
        const float x1 = acc[nc][r], x2 = acc[nc + 2][r];
        acc[nc][r]     = x1 * cs - x2 * sn;
        acc[nc + 2][r] = x1 * sn + x2 * cs;
      }
    }
  }

  #pragma unroll
  for (int r = 0; r < 8; ++r) {
    const int grow = m0 + w * 16 + r + hs * 8;
    const int npos = grow & (NSEQ - 1);
    const int batch = grow >> 11;
    const int bh = batch * HEADS + head;
    if (qkv == 2) {
      #pragma unroll
      for (int nc = 0; nc < 4; ++nc) {
        const int d = nc * 16 + nl;
        Vt[((size_t)bh * DH + d) * NSEQ + npos] = f2bf(acc[nc][r]);
      }
    } else {
      unsigned short* dst = qkv ? Kb : Qb;
      const size_t base = ((size_t)bh * NSEQ + npos) * DH;
      #pragma unroll
      for (int nc = 0; nc < 4; ++nc) {
        const int d = nc * 16 + nl;
        dst[base + d] = f2bf(acc[nc][r]);
      }
    }
  }
}

// ---------------------------------------------------------------------------
// Kernel C: flash attention per (batch*head, 64 q-rows). 128 threads, 4 waves;
// wave owns 16 q-rows. K/V tiles double-buffered in LDS, filled with
// GLOBAL_LOAD_ASYNC_TO_LDS_B128 pipelined one tile ahead (ASYNCcnt-tracked),
// so HBM/L2 latency overlaps the 16 WMMAs per tile. All WMMA operand loads
// are ds_load_b128 (fragment-native layouts). fp32 online softmax via LDS.
// ---------------------------------------------------------------------------
__global__ __launch_bounds__(128) void attn_kernel(
    const unsigned short* __restrict__ Qb, const unsigned short* __restrict__ Kb,
    const unsigned short* __restrict__ Vt, unsigned short* __restrict__ Ob) {
  __shared__ unsigned short Qs[64][80];         // 160B rows, 16B aligned
  __shared__ unsigned       Ksf[2][2048];       // fragment-native K tiles (x2)
  __shared__ unsigned       Vsf[2][2048];       // fragment-native V tiles (x2)
  __shared__ float          Ss[4][16][64];
  __shared__ unsigned short Ps[4][16][80];
  __shared__ float m_sm[64], l_sm[64], a_sm[64];

  const int tid = threadIdx.x, lane = tid & 31, w = tid >> 5;
  const int nl = lane & 15, hs = lane >> 4;
  const int q0 = blockIdx.x * 64;
  const int bh = blockIdx.y;
  const size_t hbase = (size_t)bh * NSEQ * DH;   // K base (row-major)
  const size_t vbase = (size_t)bh * DH * NSEQ;   // Vt base (transposed)
  const int mrow = w * 16 + nl;

  // per-thread staging coordinates (idx4 = tid + i*128; here precomputed parts)
  // idx4 -> j4 = idx4&1, g = idx4>>1, ls = g&31, grp = g>>5, p0 = grp>>1,
  // ks = grp&1 ; rr = p0*16 + (ls&15) ; cb = ks*32 + (ls>>4)*16 + 8*j4
  int st_rr[4], st_cb[4];
  #pragma unroll
  for (int i = 0; i < 4; ++i) {
    int idx4 = tid + i * 128;
    int j4 = idx4 & 1;
    int g = idx4 >> 1;
    int ls = g & 31, grp = g >> 5;
    st_rr[i] = (grp >> 1) * 16 + (ls & 15);
    st_cb[i] = (grp & 1) * 32 + (ls >> 4) * 16 + 8 * j4;
  }

  if (tid < 64) { m_sm[tid] = -3.0e38f; l_sm[tid] = 0.0f; }

  // stage Q tile (u32x4 copies, 4 per thread)
  #pragma unroll
  for (int i = 0; i < 4; ++i) {
    int idx4 = tid + i * 128;                 // 0..511
    int r = idx4 >> 3, c4 = idx4 & 7;
    *(u32x4*)&Qs[r][c4 * 8] =
        *(const u32x4*)(Qb + hbase + (size_t)(q0 + r) * DH + c4 * 8);
  }

  // issue async K/V loads for tile 0 into buffer 0
  #pragma unroll
  for (int i = 0; i < 4; ++i) {
    int idx4 = tid + i * 128;
    async_b128((unsigned)(size_t)&Ksf[0][idx4 * 4],
               Kb + hbase + (size_t)st_rr[i] * DH + st_cb[i]);
    async_b128((unsigned)(size_t)&Vsf[0][idx4 * 4],
               Vt + vbase + (size_t)st_rr[i] * NSEQ + st_cb[i]);
  }

  __syncthreads();   // Qs visible

  // Q A-fragments (2 x 16x32), 2x b128 each
  FragV qa[2];
  #pragma unroll
  for (int f = 0; f < 2; ++f) {
    qa[f].q[0] = *(const u32x4*)&Qs[mrow][f * 32 + hs * 16];
    qa[f].q[1] = *(const u32x4*)&Qs[mrow][f * 32 + hs * 16 + 8];
  }

  v8f o[4] = {};

  for (int t = 0; t < NSEQ / 64; ++t) {
    const int cur = t & 1;
    if (t + 1 < NSEQ / 64) {
      // issue next tile into the other buffer, then wait for current tile
      const int nxt = cur ^ 1;
      #pragma unroll
      for (int i = 0; i < 4; ++i) {
        int idx4 = tid + i * 128;
        async_b128((unsigned)(size_t)&Ksf[nxt][idx4 * 4],
                   Kb + hbase + (size_t)((t + 1) * 64 + st_rr[i]) * DH + st_cb[i]);
        async_b128((unsigned)(size_t)&Vsf[nxt][idx4 * 4],
                   Vt + vbase + (size_t)st_rr[i] * NSEQ + (t + 1) * 64 + st_cb[i]);
      }
      wait_async8();   // 8 newest still in flight; tile t complete (in-order)
    } else {
      wait_async0();
    }
    __syncthreads();   // tile t visible to all waves

    // S (16 q x 64 k) = Q Kt
    #pragma unroll
    for (int kn = 0; kn < 4; ++kn) {
      v8f s = {};
      #pragma unroll
      for (int ks = 0; ks < 2; ++ks) {
        FragV b;
        const u32x4* p =
            (const u32x4*)&Ksf[cur][(size_t)(((kn * 2 + ks) * 32 + lane) * 8)];
        b.q[0] = p[0];
        b.q[1] = p[1];
        s = wmma_bf16(qa[ks].v, b.v, s);
      }
      #pragma unroll
      for (int r = 0; r < 8; ++r)
        Ss[w][r + hs * 8][kn * 16 + nl] = s[r];
    }

    // online softmax: lanes 0..15 each own one of the wave's 16 rows
    if (lane < 16) {
      const int rloc = w * 16 + lane;
      const float mold = m_sm[rloc];
      float rmax = -3.0e38f;
      for (int jj = 0; jj < 64; ++jj)
        rmax = fmaxf(rmax, Ss[w][lane][jj] * SCALE);
      const float mnew = fmaxf(mold, rmax);
      float sum = 0.0f;
      for (int jj = 0; jj < 64; jj += 2) {
        float p0 = __expf(Ss[w][lane][jj] * SCALE - mnew);
        float p1 = __expf(Ss[w][lane][jj + 1] * SCALE - mnew);
        *(unsigned*)&Ps[w][lane][jj] = pack2(f2bf(p0), f2bf(p1));
        sum += p0 + p1;
      }
      const float alpha = __expf(mold - mnew);
      m_sm[rloc] = mnew;
      l_sm[rloc] = l_sm[rloc] * alpha + sum;
      a_sm[rloc] = alpha;
    }
    // same-wave LDS RAW: compiler orders via s_wait_dscnt

    // rescale running O by alpha (row-dependent)
    #pragma unroll
    for (int r = 0; r < 8; ++r) {
      const float al = a_sm[w * 16 + r + hs * 8];
      #pragma unroll
      for (int nc = 0; nc < 4; ++nc) o[nc][r] *= al;
    }

    // P A-fragments (2x b128 each), then O += P @ V
    FragV pa[2];
    #pragma unroll
    for (int f = 0; f < 2; ++f) {
      pa[f].q[0] = *(const u32x4*)&Ps[w][nl][f * 32 + hs * 16];
      pa[f].q[1] = *(const u32x4*)&Ps[w][nl][f * 32 + hs * 16 + 8];
    }
    #pragma unroll
    for (int nc = 0; nc < 4; ++nc) {
      #pragma unroll
      for (int ks = 0; ks < 2; ++ks) {
        FragV vb;
        const u32x4* p =
            (const u32x4*)&Vsf[cur][(size_t)(((nc * 2 + ks) * 32 + lane) * 8)];
        vb.q[0] = p[0];
        vb.q[1] = p[1];
        o[nc] = wmma_bf16(pa[ks].v, vb.v, o[nc]);
      }
    }
    __syncthreads();   // all waves done with buffer `cur` before it is refilled
  }

  // finalize: O /= l, write (b, n, h*64+d) bf16 for the output projection
  const int batch = bh >> 4, head = bh & (HEADS - 1);
  #pragma unroll
  for (int r = 0; r < 8; ++r) {
    const int rloc = w * 16 + r + hs * 8;
    const float inv = 1.0f / l_sm[rloc];
    const size_t row = (size_t)(batch * NSEQ + q0 + rloc);
    #pragma unroll
    for (int nc = 0; nc < 4; ++nc) {
      const int d = nc * 16 + nl;
      Ob[row * INNER + head * DH + d] = f2bf(o[nc][r] * inv);
    }
  }
}

// ---------------------------------------------------------------------------
// Kernel D: output projection (AttnOut[4096x1024] @ Wout[1024x1024]) + bias.
// Same tiling as kernel B; A is already bf16 (u32x4-staged), B from
// pre-swizzled global weights.
// ---------------------------------------------------------------------------
__global__ __launch_bounds__(128) void out_proj_kernel(
    const unsigned short* __restrict__ A, const unsigned short* __restrict__ Wsw,
    const float* __restrict__ bias, float* __restrict__ out) {
  __shared__ unsigned short Xs[64][40];

  const int tid = threadIdx.x, lane = tid & 31, w = tid >> 5;
  const int nl = lane & 15, hs = lane >> 4;
  const int ct = blockIdx.x;
  const int m0 = blockIdx.y * 64;
  const int mrow = w * 16 + nl;
  const u32x4* __restrict__ W4 = (const u32x4*)Wsw;

  v8f acc[4] = {};

  for (int kt = 0; kt < INNER / 32; ++kt) {
    #pragma unroll
    for (int i = 0; i < 2; ++i) {
      int idx4 = tid + i * 128;               // 0..255
      int r = idx4 >> 2, c4 = idx4 & 3;
      *(u32x4*)&Xs[r][c4 * 8] =
          *(const u32x4*)(A + (size_t)(m0 + r) * INNER + kt * 32 + c4 * 8);
    }
    __syncthreads();

    FragV a;
    a.q[0] = *(const u32x4*)&Xs[mrow][hs * 16];
    a.q[1] = *(const u32x4*)&Xs[mrow][hs * 16 + 8];

    const size_t tb = ((size_t)(ct * 32 + kt)) * 256;
    #pragma unroll
    for (int nc = 0; nc < 4; ++nc) {
      FragV b;
      const u32x4* wp = W4 + tb + (size_t)(nc * 32 + lane) * 2;
      b.q[0] = wp[0];
      b.q[1] = wp[1];
      acc[nc] = wmma_bf16(a.v, b.v, acc[nc]);
    }
    __syncthreads();
  }

  #pragma unroll
  for (int r = 0; r < 8; ++r) {
    const int grow = m0 + w * 16 + r + hs * 8;
    #pragma unroll
    for (int nc = 0; nc < 4; ++nc) {
      const int col = ct * 64 + nc * 16 + nl;
      out[(size_t)grow * DIMX + col] = acc[nc][r] + bias[col];
    }
  }
}

// ---------------------------------------------------------------------------
// Host-side launch. Workspace (bf16 stored as ushort):
//   Wqkv_sw : 1024*3072  (6 MB, fragment-swizzled)
//   Wout_sw : 1024*1024  (2 MB, fragment-swizzled)
//   Qb, Kb  : 2*16*2048*64 each (8 MB, head-major row-major)
//   Vt      : 2*16*64*2048     (8 MB, head-major transposed)
//   Ob      : 4096*1024        (8 MB)             total ~40 MB
// ---------------------------------------------------------------------------
extern "C" void kernel_launch(void* const* d_in, const int* in_sizes, int n_in,
                              void* d_out, int out_size, void* d_ws, size_t ws_size,
                              hipStream_t stream) {
  const float* x     = (const float*)d_in[0];
  const float* w_qkv = (const float*)d_in[1];
  const float* w_out = (const float*)d_in[2];
  const float* b_out = (const float*)d_in[3];
  float* out = (float*)d_out;

  const size_t headElems = (size_t)NBATCH * HEADS * NSEQ * DH;  // 4,194,304
  unsigned short* Wqkv_sw = (unsigned short*)d_ws;
  unsigned short* Wout_sw = Wqkv_sw + (size_t)DIMX * QKV3;
  unsigned short* Qb      = Wout_sw + (size_t)INNER * DIMX;
  unsigned short* Kb      = Qb + headElems;
  unsigned short* Vt      = Kb + headElems;
  unsigned short* Ob      = Vt + headElems;

  swizzle_w_kernel<<<dim3(QKV3 / 256, DIMX), 256, 0, stream>>>(w_qkv, Wqkv_sw, QKV3);
  swizzle_w_kernel<<<dim3(DIMX / 256, DIMX), 256, 0, stream>>>(w_out, Wout_sw, DIMX);

  qkv_rope_kernel<<<dim3(QKV3 / 64, NROWS / 64), 128, 0, stream>>>(
      x, Wqkv_sw, Qb, Kb, Vt);

  attn_kernel<<<dim3(NSEQ / 64, NBATCH * HEADS), 128, 0, stream>>>(
      Qb, Kb, Vt, Ob);

  out_proj_kernel<<<dim3(DIMX / 64, NROWS / 64), 128, 0, stream>>>(
      Ob, Wout_sw, b_out, out);
}
